// HetEncoder_22462678958372
// MI455X (gfx1250) — compile-verified, matrix operand order checked by reading, not run on hardware
//
#include <hip/hip_runtime.h>
#include <hip/hip_bf16.h>
#include <math.h>

// ---------------------------------------------------------------------------
// CDNA5 (gfx1250) HGT forward.
// Dense math: v_wmma_f32_16x16x32_bf16, LDS-staged tiles.
// A-tile staged via global_load_async_to_lds_b128 (ASYNCcnt), B-panel staged
// transposed so fragments load as two ds_load_b128 runs per lane.
// ---------------------------------------------------------------------------

typedef __bf16 bf16;
typedef __attribute__((ext_vector_type(16))) __bf16 v16bf;
typedef __attribute__((ext_vector_type(8)))  __bf16 v8bf;
typedef __attribute__((ext_vector_type(8)))  float  v8f;

#define ROWP 40  // LDS row pitch in bf16 elements (80B: 16B-aligned, bank-spread)

// C = gather(A)[M x K] * B[K x N] (+bias). A row-major bf16 (lda, col offset),
// optional row gather via aidx. B row-major bf16 (ldb). C f32 or bf16.
// K multiple of 32. Block: 8 waves -> 128(M) x 64(N) tile; wave w owns rows
// [w*16, w*16+16) x all 64 cols (4 WMMA accumulators).
template <bool STOREB>
__global__ __launch_bounds__(256)
void wmma_gemm_kernel(const bf16* __restrict__ A, int lda, int acol,
                      const int* __restrict__ aidx,
                      const bf16* __restrict__ B, int ldb,
                      const float* __restrict__ bias,
                      float* __restrict__ Cf, bf16* __restrict__ Cb,
                      int ldc, int ccol, int M, int K, int N, int gm) {
  __shared__ bf16 As[128 * ROWP];
  __shared__ bf16 Bs[64 * ROWP];
  __shared__ int  idx_s[128];

  int bm = (int)(blockIdx.x % (unsigned)gm);
  int bn = (int)(blockIdx.x / (unsigned)gm);
  int tid = threadIdx.x;
  int wv = tid >> 5, lane = tid & 31;
  int mlocal = lane & 15, kseg = lane >> 4, nloc = lane & 15;
  int nbase = bn * 64;

  // cache (clamped) gather indices for this block's 128 rows
  if (tid < 128) {
    int r = bm * 128 + tid;
    if (r >= M) r = M - 1;
    idx_s[tid] = aidx ? aidx[r] : r;
  }
  __syncthreads();

  v8f acc[4] = {};

  for (int k0 = 0; k0 < K; k0 += 32) {
    // ---- stage A tile (128 rows x 32 k) : async global -> LDS, b128/lane ----
    {
      int r  = tid >> 1;            // 0..127
      int c0 = (tid & 1) * 2;       // 16B chunk index: {0,1} or {2,3}
      long grow = (long)idx_s[r] * lda + acol + k0;
#pragma unroll
      for (int c = c0; c < c0 + 2; ++c) {
        const bf16* gp = A + grow + c * 8;
        unsigned lp = (unsigned)(size_t)&As[r * ROWP + c * 8];
        asm volatile("global_load_async_to_lds_b128 %0, %1, off"
                     :: "v"(lp), "v"(gp) : "memory");
      }
    }
    // ---- stage B panel transposed: Bs[n][k], one b128 load + 8 ds stores ----
    {
      int k  = tid & 31;            // k within chunk
      int nc = tid >> 5;            // 0..7 -> n = nc*8 .. +8
      const bf16* gp = B + (long)(k0 + k) * ldb + nbase + nc * 8;
      v8bf bv = *(const v8bf*)gp;
#pragma unroll
      for (int j = 0; j < 8; ++j) Bs[(nc * 8 + j) * ROWP + k] = bv[j];
    }
    asm volatile("s_wait_asynccnt 0x0" ::: "memory");
    __syncthreads();

    // ---- fragments from LDS: two contiguous 16B runs per lane (ISA 7.12.2:
    //      lane kseg half holds K in [8k,8k+8) U [16+8k,16+8k+8)) ----
    v16bf af;
    {
      const bf16* ar = &As[(wv * 16 + mlocal) * ROWP];
      v8bf lo = *(const v8bf*)&ar[8 * kseg];
      v8bf hi = *(const v8bf*)&ar[16 + 8 * kseg];
#pragma unroll
      for (int j = 0; j < 8; ++j) { af[j] = lo[j]; af[8 + j] = hi[j]; }
    }
#pragma unroll
    for (int nt = 0; nt < 4; ++nt) {
      const bf16* br = &Bs[(nt * 16 + nloc) * ROWP];
      v16bf bfr;
      v8bf lo = *(const v8bf*)&br[8 * kseg];
      v8bf hi = *(const v8bf*)&br[16 + 8 * kseg];
#pragma unroll
      for (int j = 0; j < 8; ++j) { bfr[j] = lo[j]; bfr[8 + j] = hi[j]; }
      acc[nt] = __builtin_amdgcn_wmma_f32_16x16x32_bf16(
          false, af, false, bfr, (short)0, acc[nt], false, false);
    }
    __syncthreads();
  }

  // ---- epilogue: C/D layout lanes 0-15 -> M=v, lanes 16-31 -> M=8+v ----
  int mbase = bm * 128 + wv * 16 + kseg * 8;
#pragma unroll
  for (int nt = 0; nt < 4; ++nt) {
    int col  = nbase + nt * 16 + nloc;
    float bv = bias ? bias[col] : 0.0f;
#pragma unroll
    for (int v = 0; v < 8; ++v) {
      int r = mbase + v;
      if (r < M) {
        float val = acc[nt][v] + bv;
        long o = (long)r * ldc + ccol + col;
        if (STOREB) Cb[o] = (bf16)val; else Cf[o] = val;
      }
    }
  }
}

__global__ void cast_f32_bf16_kernel(const float* __restrict__ s,
                                     bf16* __restrict__ d, long n) {
  long i = (long)blockIdx.x * blockDim.x + threadIdx.x;
  if (i < n) d[i] = (bf16)s[i];
}

__global__ void fill_u32_kernel(unsigned* __restrict__ p, unsigned v, long n) {
  long i = (long)blockIdx.x * blockDim.x + threadIdx.x;
  if (i < n) p[i] = v;
}

// LayerNorm over C channels (block == C threads, LDS tree reduction), optional
// ReLU, optional f32 and/or bf16 outputs.
__global__ void ln_act_kernel(const float* __restrict__ x,
                              const float* __restrict__ g,
                              const float* __restrict__ be,
                              float* __restrict__ outF, bf16* __restrict__ outB,
                              int C, int relu) {
  extern __shared__ float sm[];
  long row = blockIdx.x;
  int c = threadIdx.x;
  float v = x[row * C + c];
  sm[c] = v; __syncthreads();
  for (int s = C >> 1; s > 0; s >>= 1) { if (c < s) sm[c] += sm[c + s]; __syncthreads(); }
  float mu = sm[0] * (1.0f / C); __syncthreads();
  float dv = v - mu;
  sm[c] = dv * dv; __syncthreads();
  for (int s = C >> 1; s > 0; s >>= 1) { if (c < s) sm[c] += sm[c + s]; __syncthreads(); }
  float var = sm[0] * (1.0f / C);
  float y = dv * rsqrtf(var + 1e-5f) * g[c] + be[c];
  if (relu) y = fmaxf(y, 0.0f);
  if (outF) outF[row * C + c] = y;
  if (outB) outB[row * C + c] = (bf16)y;
}

// Order-preserving float <-> uint encoding for atomicMax-based segment max.
__device__ __forceinline__ unsigned encf(float f) {
  unsigned u = __float_as_uint(f);
  return (u & 0x80000000u) ? ~u : (u | 0x80000000u);
}
__device__ __forceinline__ float decf(unsigned u) {
  return __uint_as_float((u & 0x80000000u) ? (u & 0x7fffffffu) : ~u);
}

// sc[e,h] = (q[dst[e],h,:] . ke[e,h,:]) * p[h] * scale ; atomicMax into mx.
__global__ void edge_score_kernel(const float* __restrict__ ke,
                                  const float* __restrict__ q,
                                  const int* __restrict__ dsti,
                                  const float* __restrict__ pr, float scale,
                                  float* __restrict__ sc,
                                  unsigned* __restrict__ mx, int E) {
  long t = (long)blockIdx.x * blockDim.x + threadIdx.x;
  if (t >= (long)E * 4) return;
  int e = (int)(t >> 2), h = (int)(t & 3);
  int d = dsti[e];
  const float* kp = ke + (long)e * 256 + h * 64;
  const float* qp = q + (long)d * 256 + h * 64;
  float s = 0.0f;
#pragma unroll
  for (int i = 0; i < 64; ++i) s += kp[i] * qp[i];
  s *= pr[h] * scale;
  sc[t] = s;
  atomicMax(&mx[(long)d * 4 + h], encf(s));
}

// a = exp(sc - mx[dst]); agg[dst,:] += a*ve[e,:]; den[dst,h] += a.
__global__ void edge_scatter_kernel(const float* __restrict__ ve,
                                    const float* __restrict__ sc,
                                    const int* __restrict__ dsti,
                                    const unsigned* __restrict__ mx,
                                    float* __restrict__ den,
                                    float* __restrict__ agg, int E) {
  long t = (long)blockIdx.x * blockDim.x + threadIdx.x;
  if (t >= (long)E * 256) return;
  int e = (int)(t >> 8), c = (int)(t & 255), h = c >> 6;
  int d = dsti[e];
  float m = decf(mx[(long)d * 4 + h]);
  float a = __expf(sc[(long)e * 4 + h] - m);
  atomicAdd(&agg[(long)d * 256 + c], a * ve[(long)e * 256 + c]);
  if ((c & 63) == 0) atomicAdd(&den[(long)d * 4 + h], a);
}

// exact GELU of (agg / den), store bf16 for the A-projection GEMM.
__global__ void node_gelu_kernel(const float* __restrict__ agg,
                                 const float* __restrict__ den,
                                 bf16* __restrict__ out, long n) {
  long t = (long)blockIdx.x * blockDim.x + threadIdx.x;
  if (t >= n) return;
  long node = t >> 8;
  int h = (int)((t >> 6) & 3);
  float x = agg[t] / (den[node * 4 + h] + 1e-16f);
  float g = 0.5f * x * (1.0f + erff(x * 0.70710678118654752f));
  out[t] = (bf16)g;
}

// h_new = sigmoid(skip)*o + (1-sigmoid(skip))*h + h   (skip-gate + residual)
__global__ void combine_kernel(const float* __restrict__ o,
                               const float* __restrict__ skp,
                               float* __restrict__ h, bf16* __restrict__ hb,
                               long n) {
  long t = (long)blockIdx.x * blockDim.x + threadIdx.x;
  if (t >= n) return;
  float sk = 1.0f / (1.0f + __expf(-skp[0]));
  float nv = sk * o[t] + (2.0f - sk) * h[t];
  h[t] = nv;
  hb[t] = (bf16)nv;
}

extern "C" void kernel_launch(void* const* d_in, const int* in_sizes, int n_in,
                              void* d_out, int out_size, void* d_ws,
                              size_t ws_size, hipStream_t stream) {
  (void)in_sizes; (void)n_in; (void)out_size; (void)ws_size;
  // node types in data order: user, event, space, tag
  const int NN[4]  = {100000, 80000, 10000, 5000};
  const int IND[4] = {128, 256, 64, 32};
  const long Ntot  = 195000;
  long nodeOff[4]; nodeOff[0] = 0;
  for (int t = 1; t < 4; ++t) nodeOff[t] = nodeOff[t - 1] + NN[t - 1];
  // relations in data order: ue, eu, es, se, et, te, ut, tu
  const int RS[8] = {0, 1, 1, 2, 1, 3, 0, 3};
  const int RD[8] = {1, 0, 2, 1, 3, 1, 3, 0};
  const int ERN[8] = {150000, 150000, 80000, 80000, 150000, 150000, 100000, 100000};
  const int D2S[4] = {3, 0, 1, 2};              // data type -> sorted-key index
  const int R2S[8] = {6, 2, 0, 3, 1, 4, 7, 5};  // data rel  -> sorted-key index

  // --- parameter leaf indexing (jax pytree flatten: sorted dict keys) ---
  const int P0 = 20, PL = 36, PO = 156;
  auto F = [&](int i) { return (const float*)d_in[i]; };
  auto IN_B  = [&](int st) { return F(P0 + st * 4 + 0); };
  auto IN_BE = [&](int st) { return F(P0 + st * 4 + 1); };
  auto IN_G  = [&](int st) { return F(P0 + st * 4 + 2); };
  auto IN_W  = [&](int st) { return F(P0 + st * 4 + 3); };
  auto LB  = [&](int l) { return PL + l * 60; };
  auto A_B = [&](int l, int st) { return F(LB(l) + st * 2 + 0); };
  auto A_W = [&](int l, int st) { return F(LB(l) + st * 2 + 1); };
  auto K_B = [&](int l, int st) { return F(LB(l) + 8 + st * 2 + 0); };
  auto K_W = [&](int l, int st) { return F(LB(l) + 8 + st * 2 + 1); };
  auto Q_B = [&](int l, int st) { return F(LB(l) + 16 + st * 2 + 0); };
  auto Q_W = [&](int l, int st) { return F(LB(l) + 16 + st * 2 + 1); };
  auto R_A = [&](int l, int sr) { return F(LB(l) + 24 + sr * 3 + 0); };
  auto R_M = [&](int l, int sr) { return F(LB(l) + 24 + sr * 3 + 1); };
  auto R_P = [&](int l, int sr) { return F(LB(l) + 24 + sr * 3 + 2); };
  auto SKP = [&](int l, int st) { return F(LB(l) + 48 + st); };
  auto V_B = [&](int l, int st) { return F(LB(l) + 52 + st * 2 + 0); };
  auto V_W = [&](int l, int st) { return F(LB(l) + 52 + st * 2 + 1); };
  auto O_B  = [&](int st) { return F(PO + st * 4 + 0); };
  auto O_BE = [&](int st) { return F(PO + st * 4 + 1); };
  auto O_G  = [&](int st) { return F(PO + st * 4 + 2); };
  auto O_W  = [&](int st) { return F(PO + st * 4 + 3); };

  // --- workspace carve-out ---
  char* base = (char*)d_ws;
  size_t off = 0;
  auto alloc = [&](size_t bytes) -> void* {
    off = (off + 255) & ~(size_t)255;
    void* p = base + off; off += bytes; return p;
  };
  float*    h   = (float*)alloc((size_t)Ntot * 256 * 4);
  bf16*     hb  = (bf16*) alloc((size_t)Ntot * 256 * 2);
  float*    qb  = (float*)alloc((size_t)Ntot * 256 * 4);
  bf16*     kb  = (bf16*) alloc((size_t)Ntot * 256 * 2);
  bf16*     vb  = (bf16*) alloc((size_t)Ntot * 256 * 2);
  float*    agg = (float*)alloc((size_t)Ntot * 256 * 4);
  float*    den = (float*)alloc((size_t)Ntot * 4 * 4);
  unsigned* mx  = (unsigned*)alloc((size_t)Ntot * 4 * 4);
  bf16*     gel = (bf16*) alloc((size_t)Ntot * 256 * 2);
  float*    ob  = (float*)alloc((size_t)Ntot * 256 * 4);
  float*    esc = (float*)alloc((size_t)150000 * 256 * 4);
  float*    scb = (float*)alloc((size_t)960000 * 4 * 4);
  long scOff[8]; { long a = 0; for (int r = 0; r < 8; ++r) { scOff[r] = a; a += (long)ERN[r] * 4; } }
  bf16* xb = (bf16*)alloc((size_t)80000 * 256 * 2);
  bf16* winb[4]; for (int t = 0; t < 4; ++t) winb[t] = (bf16*)alloc((size_t)IND[t] * 256 * 2);
  bf16 *wkb[2][4], *wqb[2][4], *wvb[2][4], *wab[2][4];
  for (int l = 0; l < 2; ++l)
    for (int t = 0; t < 4; ++t) {
      wkb[l][t] = (bf16*)alloc(65536 * 2); wqb[l][t] = (bf16*)alloc(65536 * 2);
      wvb[l][t] = (bf16*)alloc(65536 * 2); wab[l][t] = (bf16*)alloc(65536 * 2);
    }
  bf16 *rab[2][8], *rmb[2][8];
  for (int l = 0; l < 2; ++l)
    for (int r = 0; r < 8; ++r) {
      rab[l][r] = (bf16*)alloc(16384 * 2); rmb[l][r] = (bf16*)alloc(16384 * 2);
    }
  bf16* wob[4]; for (int t = 0; t < 4; ++t) wob[t] = (bf16*)alloc(32768 * 2);

  auto cast = [&](const float* s, bf16* d, long n) {
    unsigned b = (unsigned)((n + 255) / 256);
    cast_f32_bf16_kernel<<<b, 256, 0, stream>>>(s, d, n);
  };
  auto fill = [&](unsigned* p, unsigned v, long n) {
    unsigned b = (unsigned)((n + 255) / 256);
    fill_u32_kernel<<<b, 256, 0, stream>>>(p, v, n);
  };
  auto gemm = [&](const bf16* A, int lda, int acol, const int* aidx,
                  const bf16* B, int ldb, const float* bias, float* Cf,
                  bf16* Cb, int ldc, int ccol, int M, int K, int N) {
    int gm = (M + 127) / 128;
    int gn = N / 64;
    unsigned blocks = (unsigned)(gm * gn);
    if (Cb) wmma_gemm_kernel<true ><<<blocks, 256, 0, stream>>>(A, lda, acol, aidx, B, ldb, bias, nullptr, Cb, ldc, ccol, M, K, N, gm);
    else    wmma_gemm_kernel<false><<<blocks, 256, 0, stream>>>(A, lda, acol, aidx, B, ldb, bias, Cf, nullptr, ldc, ccol, M, K, N, gm);
  };

  // --- cast all weights to bf16 (recast every call: deterministic) ---
  for (int t = 0; t < 4; ++t) cast(IN_W(D2S[t]), winb[t], (long)IND[t] * 256);
  for (int l = 0; l < 2; ++l)
    for (int t = 0; t < 4; ++t) {
      cast(K_W(l, D2S[t]), wkb[l][t], 65536); cast(Q_W(l, D2S[t]), wqb[l][t], 65536);
      cast(V_W(l, D2S[t]), wvb[l][t], 65536); cast(A_W(l, D2S[t]), wab[l][t], 65536);
    }
  for (int l = 0; l < 2; ++l)
    for (int r = 0; r < 8; ++r) {
      cast(R_A(l, R2S[r]), rab[l][r], 16384); cast(R_M(l, R2S[r]), rmb[l][r], 16384);
    }
  for (int t = 0; t < 4; ++t) cast(O_W(D2S[t]), wob[t], 32768);

  // --- input projection + LN + ReLU ---
  for (int t = 0; t < 4; ++t) {
    cast((const float*)d_in[t], xb, (long)NN[t] * IND[t]);
    gemm(xb, IND[t], 0, nullptr, winb[t], 256, IN_B(D2S[t]),
         ob + nodeOff[t] * 256, nullptr, 256, 0, NN[t], IND[t], 256);
    ln_act_kernel<<<NN[t], 256, 256 * 4, stream>>>(
        ob + nodeOff[t] * 256, IN_G(D2S[t]), IN_BE(D2S[t]),
        h + nodeOff[t] * 256, hb + nodeOff[t] * 256, 256, 1);
  }

  // --- HGT layers ---
  for (int l = 0; l < 2; ++l) {
    for (int t = 0; t < 4; ++t) {
      gemm(hb + nodeOff[t] * 256, 256, 0, nullptr, wkb[l][t], 256, K_B(l, D2S[t]),
           nullptr, kb + nodeOff[t] * 256, 256, 0, NN[t], 256, 256);
      gemm(hb + nodeOff[t] * 256, 256, 0, nullptr, wqb[l][t], 256, Q_B(l, D2S[t]),
           qb + nodeOff[t] * 256, nullptr, 256, 0, NN[t], 256, 256);
      gemm(hb + nodeOff[t] * 256, 256, 0, nullptr, wvb[l][t], 256, V_B(l, D2S[t]),
           nullptr, vb + nodeOff[t] * 256, 256, 0, NN[t], 256, 256);
    }
    fill(mx, 0u, Ntot * 4);
    fill((unsigned*)den, 0u, Ntot * 4);
    fill((unsigned*)agg, 0u, Ntot * 256);
    // pass 1: scores + segment max
    for (int r = 0; r < 8; ++r) {
      const int* src = (const int*)d_in[4 + 2 * r];
      const int* dst = (const int*)d_in[5 + 2 * r];
      int st = RS[r], dt = RD[r];
      for (int hh = 0; hh < 4; ++hh)  // ke[e,h,:] = k[src[e],h,:] @ relA[h]
        gemm(kb + nodeOff[st] * 256, 256, hh * 64, src, rab[l][r] + hh * 4096, 64,
             nullptr, esc, nullptr, 256, hh * 64, ERN[r], 64, 64);
      unsigned b = (unsigned)(((long)ERN[r] * 4 + 255) / 256);
      edge_score_kernel<<<b, 256, 0, stream>>>(
          esc, qb + nodeOff[dt] * 256, dst, R_P(l, R2S[r]), 0.125f,
          scb + scOff[r], mx + nodeOff[dt] * 4, ERN[r]);
    }
    // pass 2: exp + weighted scatter of messages
    for (int r = 0; r < 8; ++r) {
      const int* src = (const int*)d_in[4 + 2 * r];
      const int* dst = (const int*)d_in[5 + 2 * r];
      int st = RS[r], dt = RD[r];
      for (int hh = 0; hh < 4; ++hh)  // ve[e,h,:] = v[src[e],h,:] @ relM[h]
        gemm(vb + nodeOff[st] * 256, 256, hh * 64, src, rmb[l][r] + hh * 4096, 64,
             nullptr, esc, nullptr, 256, hh * 64, ERN[r], 64, 64);
      unsigned b = (unsigned)(((long)ERN[r] * 256 + 255) / 256);
      edge_scatter_kernel<<<b, 256, 0, stream>>>(
          esc, scb + scOff[r], dst, mx + nodeOff[dt] * 4,
          den + nodeOff[dt] * 4, agg + nodeOff[dt] * 256, ERN[r]);
    }
    {
      unsigned b = (unsigned)((Ntot * 256 + 255) / 256);
      node_gelu_kernel<<<b, 256, 0, stream>>>(agg, den, gel, Ntot * 256);
    }
    for (int t = 0; t < 4; ++t) {
      gemm(gel + nodeOff[t] * 256, 256, 0, nullptr, wab[l][t], 256, A_B(l, D2S[t]),
           ob + nodeOff[t] * 256, nullptr, 256, 0, NN[t], 256, 256);
      unsigned b = (unsigned)(((long)NN[t] * 256 + 255) / 256);
      combine_kernel<<<b, 256, 0, stream>>>(
          ob + nodeOff[t] * 256, SKP(l, D2S[t]), h + nodeOff[t] * 256,
          hb + nodeOff[t] * 256, (long)NN[t] * 256);
    }
  }

  // --- output projection + LN -> d_out (user, event, space, tag concat) ---
  float* out = (float*)d_out;
  for (int t = 0; t < 4; ++t) {
    gemm(hb + nodeOff[t] * 256, 256, 0, nullptr, wob[t], 128, O_B(D2S[t]),
         ob + nodeOff[t] * 128, nullptr, 128, 0, NN[t], 256, 128);
    ln_act_kernel<<<NN[t], 128, 128 * 4, stream>>>(
        ob + nodeOff[t] * 128, O_G(D2S[t]), O_BE(D2S[t]),
        out + nodeOff[t] * 128, nullptr, 128, 0);
  }
}
